// PDCM_54812372631640
// MI455X (gfx1250) — compile-verified, matrix operand order checked by reading, not run on hardware
//
#include <hip/hip_runtime.h>
#include <hip/hip_bf16.h>

// ---------------------------------------------------------------------------
// Deformable-conv alignment (3 stages) for MI455X (gfx1250, wave32).
// GEMMs on v_wmma_f32_16x16x32_bf16 (f32 accumulate); bilinear sampling in f32
// VALU (co-executes with 16-bit XDL WMMA per CDNA5 ISA §8.4). B-panels staged
// in LDS per block; A-fragments built branch-free (clamped loads + cndmask).
// ---------------------------------------------------------------------------

#define BN 4
#define CIN 64
#define OCH 64
#define HH 128
#define WW 128
#define HWSZ (HH * WW)
#define PANEL_U16 36864      // u16 elements per weight set (73728 B, both shapes)

typedef __attribute__((ext_vector_type(16))) __bf16 bf16x16;
typedef __attribute__((ext_vector_type(8)))  float  f32x8;

// CDNA5 16-bit A/B fragment K mapping (ISA 7.12.2), half-independent part:
//  v0..3 -> K {0,2,4,6}+..., v4..7 -> K {16,18,20,22}+...; +8 for lanes 16-31.
__device__ __forceinline__ constexpr int kv_of(int v) {
    return (v & 4) ? (16 + ((v & 3) << 1)) : (v << 1);
}

// ---------------------------------------------------------------------------
// Weight repack: f32 (O, Cin, 3, 3) -> bf16 B-panels, K order k = tap*Cin + c.
// Panel (kt, nt) = 32 lanes x 16 bf16 (one contiguous 32B read per lane).
// ---------------------------------------------------------------------------
__global__ void pack_panels(const float* __restrict__ w,
                            unsigned short* __restrict__ dst,
                            int Cin, int Nout, int Ntiles, int total) {
    int tid = blockIdx.x * blockDim.x + threadIdx.x;
    if (tid >= total) return;
    int s    = tid & 15;
    int lane = (tid >> 4) & 31;
    int pid  = tid >> 9;          // panel id = kt*Ntiles + nt
    int nt   = pid % Ntiles;
    int kt   = pid / Ntiles;
    int v = s >> 1, odd = s & 1;
    int half = lane >> 4, n = lane & 15;
    int k   = kt * 32 + kv_of(v) + (half << 3) + odd;
    int tap = k / Cin;
    int c   = k % Cin;
    int o   = nt * 16 + n;
    float val = (o < Nout) ? w[(o * Cin + c) * 9 + tap] : 0.0f;
    __bf16 h = (__bf16)val;
    dst[(size_t)pid * 512 + lane * 16 + s] = __builtin_bit_cast(unsigned short, h);
}

// ---------------------------------------------------------------------------
// Offset conv: 3x3, Cin=128 (implicit concat[ref, prev]), Cout=18 (padded 32).
// Block = 4 waves; wave w handles pixel tile blockIdx.x*4 + w. B-panels in LDS.
// K = 1152: tap-major -> chunk = (tap, 32-ch slice); slices 0-1 = ref, 2-3 = prev
// (compile-time base select after unrolling).
// ---------------------------------------------------------------------------
__global__ void __launch_bounds__(128)
offset_conv(const float* __restrict__ ref, const float* __restrict__ prev,
            const unsigned short* __restrict__ panels_u16,
            const float* __restrict__ bias, float* __restrict__ out) {
    __shared__ __align__(32) unsigned short s_panels[PANEL_U16];
    {
        const int t = threadIdx.y * 32 + threadIdx.x;      // 0..127
        const uint4* src = (const uint4*)panels_u16;
        uint4* dst = (uint4*)s_panels;
#pragma unroll
        for (int i = 0; i < 36; ++i)                       // 4608 uint4 / 128 thr
            dst[t + i * 128] = src[t + i * 128];
    }
    __syncthreads();
    const __bf16* span = reinterpret_cast<const __bf16*>(s_panels);

    const int lane = threadIdx.x;
    const int m = lane & 15, half = lane >> 4;
    const int xt = blockIdx.x * 4 + threadIdx.y;           // 16-pixel tile index
    const int x = xt * 16 + m;
    const int y = blockIdx.y, b = blockIdx.z;
    // fold the half*8 channel offset into the bases once
    const float* refb  = ref  + (size_t)b * CIN * HWSZ + (size_t)half * 8 * HWSZ;
    const float* prevb = prev + (size_t)b * CIN * HWSZ + (size_t)half * 8 * HWSZ;

    f32x8 acc0 = {}, acc1 = {};
    for (int tap = 0; tap < 9; ++tap) {
        const int ty = tap / 3;
        const int dy = ty - 1, dx = (tap - ty * 3) - 1;
        const int yy = y + dy, xx = x + dx;
        const bool inb = ((unsigned)yy < HH) && ((unsigned)xx < WW);
        const int yyc = min(max(yy, 0), HH - 1);
        const int xxc = min(max(xx, 0), WW - 1);
        const int sp = yyc * WW + xxc;                     // always in-bounds

#pragma unroll
        for (int ct = 0; ct < 4; ++ct) {                   // 32-ch slices of concat
            const float* basep = ((ct < 2) ? refb : prevb) + sp;   // compile-time sel
            const int c0 = (ct & 1) << 5;
            bf16x16 a;
#pragma unroll
            for (int v = 0; v < 8; ++v) {
                const size_t co = (size_t)(c0 + kv_of(v)) * HWSZ;
                const float f0 = basep[co];                // unconditional loads
                const float f1 = basep[co + HWSZ];
                a[2 * v]     = (__bf16)(inb ? f0 : 0.f);   // v_cndmask, no branches
                a[2 * v + 1] = (__bf16)(inb ? f1 : 0.f);
            }
            const int kt = tap * 4 + ct;
            const bf16x16 b0 = *(const bf16x16*)(span + (size_t)(kt * 2 + 0) * 512 + lane * 16);
            const bf16x16 b1 = *(const bf16x16*)(span + (size_t)(kt * 2 + 1) * 512 + lane * 16);
            acc0 = __builtin_amdgcn_wmma_f32_16x16x32_bf16(false, a, false, b0, (short)0, acc0, false, false);
            acc1 = __builtin_amdgcn_wmma_f32_16x16x32_bf16(false, a, false, b1, (short)0, acc1, false, false);
        }
    }

    // C/D layout: lane n = lane&15 -> column N; VGPR r -> row M = r + 8*half.
    const int n = m;
    const float b0v = bias[n];
    const bool hi = (n < 2);
    const float b1v = hi ? bias[16 + n] : 0.f;
#pragma unroll
    for (int r = 0; r < 8; ++r) {
        const int xp = xt * 16 + r + half * 8;
        out[(((size_t)b * 18 + n) * HH + y) * WW + xp] = acc0[r] + b0v;
        if (hi)
            out[(((size_t)b * 18 + 16 + n) * HH + y) * WW + xp] = acc1[r] + b1v;
    }
}

// ---------------------------------------------------------------------------
// Deformable conv: bilinear sample 64 ch x 9 taps (f32 VALU, corner-clamped,
// weight-masked -> branch-free), contract with bf16 WMMA. K = 576 -> 18 chunks,
// N = 64 -> 4 tiles. Block = 4 waves, B-panels in LDS.
// ---------------------------------------------------------------------------
__global__ void __launch_bounds__(128)
deform_conv(const float* __restrict__ xin, const float* __restrict__ off,
            const unsigned short* __restrict__ panels_u16,
            const float* __restrict__ bias, float* __restrict__ out) {
    __shared__ __align__(32) unsigned short s_panels[PANEL_U16];
    {
        const int t = threadIdx.y * 32 + threadIdx.x;
        const uint4* src = (const uint4*)panels_u16;
        uint4* dst = (uint4*)s_panels;
#pragma unroll
        for (int i = 0; i < 36; ++i)
            dst[t + i * 128] = src[t + i * 128];
    }
    __syncthreads();
    const __bf16* span = reinterpret_cast<const __bf16*>(s_panels);

    const int lane = threadIdx.x;
    const int m = lane & 15, half = lane >> 4;
    const int xt = blockIdx.x * 4 + threadIdx.y;
    const int x = xt * 16 + m;
    const int y = blockIdx.y, b = blockIdx.z;
    const float* xb   = xin + (size_t)b * CIN * HWSZ + (size_t)half * 8 * HWSZ;
    const float* offb = off + (size_t)b * 18 * HWSZ + y * WW + x;

    f32x8 acc[4] = {{}, {}, {}, {}};
    for (int tap = 0; tap < 9; ++tap) {
        const int ty = tap / 3;
        // offset tensor is (B, 9, 2, H, W)
        const float oy = offb[(size_t)(tap * 2 + 0) * HWSZ];
        const float ox = offb[(size_t)(tap * 2 + 1) * HWSZ];
        const float py = (float)(y + ty - 1) + oy;
        const float px = (float)(x + (tap - ty * 3) - 1) + ox;
        const float fy = floorf(py), fx = floorf(px);
        const float ly = py - fy, lx = px - fx;
        const int y0 = (int)fy, x0 = (int)fx;
        const int y1 = y0 + 1,  x1 = x0 + 1;
        const bool vy0 = (unsigned)y0 < HH, vy1 = (unsigned)y1 < HH;
        const bool vx0 = (unsigned)x0 < WW, vx1 = (unsigned)x1 < WW;
        const int cy0 = min(max(y0, 0), HH - 1), cy1 = min(max(y1, 0), HH - 1);
        const int cx0 = min(max(x0, 0), WW - 1), cx1 = min(max(x1, 0), WW - 1);
        const int i00 = cy0 * WW + cx0, i01 = cy0 * WW + cx1;
        const int i10 = cy1 * WW + cx0, i11 = cy1 * WW + cx1;
        const float w00 = (vy0 && vx0) ? (1.f - ly) * (1.f - lx) : 0.f;
        const float w01 = (vy0 && vx1) ? (1.f - ly) * lx         : 0.f;
        const float w10 = (vy1 && vx0) ? ly * (1.f - lx)         : 0.f;
        const float w11 = (vy1 && vx1) ? ly * lx                 : 0.f;

#pragma unroll
        for (int ct = 0; ct < 2; ++ct) {
            const int kt = tap * 2 + ct;
            bf16x16 a;
#pragma unroll
            for (int v = 0; v < 8; ++v) {
                const float* p0 = xb + (size_t)((ct << 5) + kv_of(v)) * HWSZ;
                const float* p1 = p0 + HWSZ;
                const float s0 = p0[i00] * w00 + p0[i01] * w01 + p0[i10] * w10 + p0[i11] * w11;
                const float s1 = p1[i00] * w00 + p1[i01] * w01 + p1[i10] * w10 + p1[i11] * w11;
                a[2 * v]     = (__bf16)s0;
                a[2 * v + 1] = (__bf16)s1;
            }
#pragma unroll
            for (int nt = 0; nt < 4; ++nt) {
                const bf16x16 bp = *(const bf16x16*)(span + (size_t)(kt * 4 + nt) * 512 + lane * 16);
                acc[nt] = __builtin_amdgcn_wmma_f32_16x16x32_bf16(false, a, false, bp, (short)0, acc[nt], false, false);
            }
        }
    }

    const int n = m;
#pragma unroll
    for (int nt = 0; nt < 4; ++nt) {
        const int o = nt * 16 + n;
        const float bo = bias[o];
#pragma unroll
        for (int r = 0; r < 8; ++r) {
            const int xp = xt * 16 + r + half * 8;
            out[(((size_t)b * OCH + o) * HH + y) * WW + xp] = acc[nt][r] + bo;
        }
    }
}

// ---------------------------------------------------------------------------
// Launch: pack 6 weight sets, then offset1 -> d1 -> offset2 -> d2 -> offset3 -> d3
// ---------------------------------------------------------------------------
extern "C" void kernel_launch(void* const* d_in, const int* in_sizes, int n_in,
                              void* d_out, int out_size, void* d_ws, size_t ws_size,
                              hipStream_t stream) {
    const float* ref    = (const float*)d_in[0];
    const float* nbr    = (const float*)d_in[1];
    const float* w_off1 = (const float*)d_in[2];
    const float* b_off1 = (const float*)d_in[3];
    const float* w_d1   = (const float*)d_in[4];
    const float* b_d1   = (const float*)d_in[5];
    const float* w_off2 = (const float*)d_in[6];
    const float* b_off2 = (const float*)d_in[7];
    const float* w_d2   = (const float*)d_in[8];
    const float* b_d2   = (const float*)d_in[9];
    const float* w_off3 = (const float*)d_in[10];
    const float* b_off3 = (const float*)d_in[11];
    const float* w_d3   = (const float*)d_in[12];
    const float* b_d3   = (const float*)d_in[13];

    char* ws = (char*)d_ws;
    const size_t PANEL_BYTES = (size_t)PANEL_U16 * 2;      // 73728 B per set
    unsigned short* pw1 = (unsigned short*)(ws + 0 * PANEL_BYTES);
    unsigned short* pw2 = (unsigned short*)(ws + 1 * PANEL_BYTES);
    unsigned short* pw3 = (unsigned short*)(ws + 2 * PANEL_BYTES);
    unsigned short* pd1 = (unsigned short*)(ws + 3 * PANEL_BYTES);
    unsigned short* pd2 = (unsigned short*)(ws + 4 * PANEL_BYTES);
    unsigned short* pd3 = (unsigned short*)(ws + 5 * PANEL_BYTES);
    float* offbuf = (float*)(ws + 6 * PANEL_BYTES);                       // 4.7 MB
    float* feat1  = (float*)(ws + 6 * PANEL_BYTES + 4718592ull);          // 16.8 MB
    float* feat2  = (float*)(ws + 6 * PANEL_BYTES + 4718592ull + 16777216ull);

    const int packTotal  = PANEL_U16;                      // Ktiles*Ntiles*512
    const int packBlocks = (packTotal + 255) / 256;
    pack_panels<<<packBlocks, 256, 0, stream>>>(w_off1, pw1, 128, 18, 2, packTotal);
    pack_panels<<<packBlocks, 256, 0, stream>>>(w_off2, pw2, 128, 18, 2, packTotal);
    pack_panels<<<packBlocks, 256, 0, stream>>>(w_off3, pw3, 128, 18, 2, packTotal);
    pack_panels<<<packBlocks, 256, 0, stream>>>(w_d1,   pd1,  64, 64, 4, packTotal);
    pack_panels<<<packBlocks, 256, 0, stream>>>(w_d2,   pd2,  64, 64, 4, packTotal);
    pack_panels<<<packBlocks, 256, 0, stream>>>(w_d3,   pd3,  64, 64, 4, packTotal);

    dim3 grid(WW / 64, HH, BN);     // 2 x 128 x 4 blocks
    dim3 blk(32, 4);                // 4 wave32 per block, one pixel tile each
    offset_conv<<<grid, blk, 0, stream>>>(ref, nbr,   pw1, b_off1, offbuf);
    deform_conv<<<grid, blk, 0, stream>>>(nbr,  offbuf, pd1, b_d1, feat1);
    offset_conv<<<grid, blk, 0, stream>>>(ref, feat1, pw2, b_off2, offbuf);
    deform_conv<<<grid, blk, 0, stream>>>(feat1, offbuf, pd2, b_d2, feat2);
    offset_conv<<<grid, blk, 0, stream>>>(ref, feat2, pw3, b_off3, offbuf);
    deform_conv<<<grid, blk, 0, stream>>>(feat2, offbuf, pd3, b_d3, (float*)d_out);
}
